// MultiheadAttention_46394236731800
// MI455X (gfx1250) — matrix operand order repaired and checked by reading, not compiled, not run on hardware
//
#include <hip/hip_runtime.h>
#include <hip/hip_bf16.h>
#include <stdint.h>

#define TGT 2048
#define BSZ 4
#define NH  16
#define HD  64
#define EMB 1024
#define PLD 2064   // padded bf16 row stride for P in LDS (kills ds bank conflicts)

typedef __attribute__((ext_vector_type(16))) __bf16 bf16x16;
typedef __attribute__((ext_vector_type(8)))  float  f32x8;

union FragU { uint4 q[2]; bf16x16 v; };

__device__ __forceinline__ f32x8 wmma_bf16(bf16x16 a, bf16x16 b, f32x8 c) {
  // D = A(16x32 bf16) * B(32x16 bf16) + C(16x16 f32)
  return __builtin_amdgcn_wmma_f32_16x16x32_bf16(false, a, false, b, (short)0, c,
                                                 false, false);
}

// A-matrix 16x32 bf16 fragment: lane = M (mod 16); K chunks {ko..ko+7, ko+16..ko+23},
// ko = (lane>>4)*8  (per ISA 7.12.2 table)
__device__ __forceinline__ bf16x16 load_frag_a(const __hip_bfloat16* base, int ld,
                                               int m0, int k0, int lane) {
  const __hip_bfloat16* p = base + (size_t)(m0 + (lane & 15)) * ld + k0 + ((lane >> 4) << 3);
  FragU f;
  f.q[0] = *reinterpret_cast<const uint4*>(p);
  f.q[1] = *reinterpret_cast<const uint4*>(p + 16);
  return f.v;
}

// B-matrix 32x16 bf16 fragment: lane = N (mod 16); K chunks {ko..ko+7, ko+16..ko+23},
// ko = (lane>>4)*8  (same K order as the A fragment)
__device__ __forceinline__ bf16x16 load_frag_b(const __hip_bfloat16* base, int ld,
                                               int n0, int k0, int lane) {
  const __hip_bfloat16* p = base + (size_t)(n0 + (lane & 15)) * ld + k0 + ((lane >> 4) << 3);
  FragU f;
  f.q[0] = *reinterpret_cast<const uint4*>(p);
  f.q[1] = *reinterpret_cast<const uint4*>(p + 16);
  return f.v;
}

// ---------------- fp32 -> bf16 conversion ----------------
__global__ void cvt_f32_bf16(const float* __restrict__ src,
                             __hip_bfloat16* __restrict__ dst, int n) {
  for (int i = blockIdx.x * blockDim.x + threadIdx.x; i < n;
       i += gridDim.x * blockDim.x)
    dst[i] = __float2bfloat16(src[i]);
}

// ---------------- GEMM: C[M,N] = (A[M,K] @ W[N,K]^T + bias) * scale ----------------
// MODE 0: store bf16 row-major [M,N]
// MODE 1: store bf16 scattered into vt[b][h][d][t]   (V projection)
// MODE 2: store f32 row-major [M,N]                  (output projection)
// Software-pipelined: next k-step's fragments are prefetched before the current
// step's WMMAs so loads overlap the matrix pipe instead of s_wait_loadcnt 0 stalls.
template <int MODE>
__global__ __launch_bounds__(256) void gemm_wmma(
    const __hip_bfloat16* __restrict__ A,
    const __hip_bfloat16* __restrict__ W,
    const float* __restrict__ bias,
    void* __restrict__ Cout,
    int M, int N, int K, float scale)
{
  const int lane = threadIdx.x & 31;
  const int wv   = threadIdx.x >> 5;                 // 8 waves
  const int m0   = blockIdx.y * 64  + (wv & 3) * 16; // 64-row block tile
  const int n0   = blockIdx.x * 128 + (wv >> 2) * 64;// 128-col block tile

  f32x8 acc[4] = {};
  bf16x16 a_cur = load_frag_a(A, K, m0, 0, lane);
  bf16x16 b_cur[4];
#pragma unroll
  for (int j = 0; j < 4; ++j) b_cur[j] = load_frag_b(W, K, n0 + j * 16, 0, lane);

  for (int k = 32; k < K; k += 32) {
    bf16x16 a_nxt = load_frag_a(A, K, m0, k, lane);  // prefetch k-step
    bf16x16 b_nxt[4];
#pragma unroll
    for (int j = 0; j < 4; ++j) b_nxt[j] = load_frag_b(W, K, n0 + j * 16, k, lane);
#pragma unroll
    for (int j = 0; j < 4; ++j) acc[j] = wmma_bf16(a_cur, b_cur[j], acc[j]);
    a_cur = a_nxt;
#pragma unroll
    for (int j = 0; j < 4; ++j) b_cur[j] = b_nxt[j];
  }
#pragma unroll
  for (int j = 0; j < 4; ++j) acc[j] = wmma_bf16(a_cur, b_cur[j], acc[j]);

  const int cn = lane & 15, rb = (lane >> 4) << 3;
#pragma unroll
  for (int j = 0; j < 4; ++j) {
#pragma unroll
    for (int r = 0; r < 8; ++r) {
      int mm = m0 + rb + r;
      int nn = n0 + j * 16 + cn;
      float v = (acc[j][r] + bias[nn]) * scale;
      if (MODE == 0) {
        ((__hip_bfloat16*)Cout)[(size_t)mm * N + nn] = __float2bfloat16(v);
      } else if (MODE == 1) {
        int t = mm >> 2, b2 = mm & 3;     // m = t*BSZ + b
        int h = nn >> 6, d  = nn & 63;    // n = h*HD + d
        ((__hip_bfloat16*)Cout)[((size_t)(b2 * NH + h) * HD + d) * TGT + t] =
            __float2bfloat16(v);
      } else {
        ((float*)Cout)[(size_t)mm * N + nn] = v;
      }
    }
  }
}

// ---------------- fused attention: scores + softmax + avg + P*V ----------------
// One workgroup owns (batch b, 16 query rows); loops over 16 heads.
// LDS: S fp32 [16][2048] | P bf16 [16][PLD] | part fp32 [2][16][64]
__global__ __launch_bounds__(256) void attn_wmma(
    const __hip_bfloat16* __restrict__ qb,   // [T,B,E] (pre-scaled)
    const __hip_bfloat16* __restrict__ kb,   // [T,B,E]
    const __hip_bfloat16* __restrict__ vt,   // [B,H,HD,T]
    __hip_bfloat16* __restrict__ ctx,        // [T,B,E]
    float* __restrict__ avg)                 // [B,T,T]  (accumulated in place)
{
  extern __shared__ char smem[];
  float*          S    = (float*)smem;                               // 128 KB
  __hip_bfloat16* P    = (__hip_bfloat16*)(smem + 131072);           //  ~64 KB
  float*          part = (float*)(smem + 131072 + 16 * PLD * 2);     //   8 KB

  const int tid  = threadIdx.x;
  const int lane = tid & 31;
  const int wv   = tid >> 5;                 // 8 waves
  const int b    = blockIdx.x & 3;
  const int q0   = (blockIdx.x >> 2) << 4;   // 16 query rows per block

  for (int h = 0; h < NH; ++h) {
    // -------- phase 1: S[16,2048] = Qh @ Kh^T (WMMA, K-frag double-buffered) ----
    const __hip_bfloat16* qbase = qb + (size_t)b * EMB + h * HD;
    const __hip_bfloat16* kbase = kb + (size_t)b * EMB + h * HD;
    bf16x16 a0 = load_frag_a(qbase, BSZ * EMB, q0, 0,  lane);  // HD=64 -> 2 K-steps
    bf16x16 a1 = load_frag_a(qbase, BSZ * EMB, q0, 32, lane);
    const int tk0 = wv * 256;                                  // each wave: 256 keys
    bf16x16 kb0 = load_frag_b(kbase, BSZ * EMB, tk0, 0,  lane);
    bf16x16 kb1 = load_frag_b(kbase, BSZ * EMB, tk0, 32, lane);
    for (int tt = 0; tt < 256; tt += 16) {
      bf16x16 nb0, nb1;
      if (tt + 16 < 256) {                                     // prefetch next tile
        nb0 = load_frag_b(kbase, BSZ * EMB, tk0 + tt + 16, 0,  lane);
        nb1 = load_frag_b(kbase, BSZ * EMB, tk0 + tt + 16, 32, lane);
      }
      f32x8 c = {};
      c = wmma_bf16(a0, kb0, c);
      c = wmma_bf16(a1, kb1, c);
      int cn = tk0 + tt + (lane & 15), rb = (lane >> 4) << 3;
#pragma unroll
      for (int r = 0; r < 8; ++r) S[(rb + r) * 2048 + cn] = c[r];
      if (tt + 16 < 256) { kb0 = nb0; kb1 = nb1; }
    }
    __syncthreads();

    // -------- phase 2: softmax row-wise; P=bf16; accumulate head-avg --------
    {
      const int r = tid >> 4, cl = tid & 15;   // 16 threads per row
      float* Srow = S + r * 2048;
      float mx = -3.0e38f;
      for (int j = 0; j < 128; ++j) mx = fmaxf(mx, Srow[cl + j * 16]);
      for (int msk = 1; msk < 16; msk <<= 1) mx = fmaxf(mx, __shfl_xor(mx, msk, 32));
      float sum = 0.f;
      for (int j = 0; j < 128; ++j) {
        float e = __expf(Srow[cl + j * 16] - mx);
        Srow[cl + j * 16] = e;
        sum += e;
      }
      for (int msk = 1; msk < 16; msk <<= 1) sum += __shfl_xor(sum, msk, 32);
      const float inv = 1.0f / sum;
      float* avgrow = avg + ((size_t)b * TGT + q0 + r) * TGT;  // exclusively owned
      __hip_bfloat16* Prow = P + r * PLD;
      for (int j = 0; j < 128; ++j) {
        int cc = cl + j * 16;
        float p = Srow[cc] * inv;
        Prow[cc] = __float2bfloat16(p);
        if (h == 0) avgrow[cc] = p;
        else {
          float a2 = avgrow[cc] + p;
          avgrow[cc] = (h == NH - 1) ? a2 * (1.0f / NH) : a2;
        }
      }
    }
    __syncthreads();

    // -------- phase 3: ctx[16,64] = P @ V  (V pre-transposed: rows = d) --------
    {
      const int nt = wv & 3, khalf = wv >> 2;  // 4 d-tiles x 2 key-halves
      const __hip_bfloat16* vb = vt + (size_t)(b * NH + h) * HD * TGT;
      f32x8 c = {};
      const int kbeg = khalf * 1024, kend = kbeg + 1024;
      bf16x16 vb_cur = load_frag_b(vb, TGT, nt * 16, kbeg, lane);
      for (int kk = kbeg; kk < kend; kk += 32) {
        bf16x16 vb_nxt;
        if (kk + 32 < kend)                              // prefetch next V fragment
          vb_nxt = load_frag_b(vb, TGT, nt * 16, kk + 32, lane);
        const __hip_bfloat16* pp = P + (lane & 15) * PLD + kk + ((lane >> 4) << 3);
        FragU f;                                  // A-fragment from LDS (ds_load_b128)
        f.q[0] = *reinterpret_cast<const uint4*>(pp);
        f.q[1] = *reinterpret_cast<const uint4*>(pp + 16);
        c = wmma_bf16(f.v, vb_cur, c);
        if (kk + 32 < kend) vb_cur = vb_nxt;
      }
      int cn = nt * 16 + (lane & 15), rb2 = (lane >> 4) << 3;
#pragma unroll
      for (int r = 0; r < 8; ++r)
        part[(khalf * 16 + rb2 + r) * 64 + cn] = c[r];
    }
    __syncthreads();
    for (int i = tid; i < 16 * 64; i += 256) {   // reduce 2 key-halves, emit bf16 ctx
      int rr = i >> 6, dd = i & 63;
      float v = part[rr * 64 + dd] + part[(16 + rr) * 64 + dd];
      ctx[((size_t)(q0 + rr) * BSZ + b) * EMB + h * HD + dd] = __float2bfloat16(v);
    }
    __syncthreads();
  }
}

extern "C" void kernel_launch(void* const* d_in, const int* in_sizes, int n_in,
                              void* d_out, int out_size, void* d_ws, size_t ws_size,
                              hipStream_t stream) {
  (void)in_sizes; (void)n_in; (void)out_size; (void)ws_size;
  const float* x  = (const float*)d_in[0];
  const float* wq = (const float*)d_in[1];
  const float* bq = (const float*)d_in[2];
  const float* wk = (const float*)d_in[3];
  const float* bk = (const float*)d_in[4];
  const float* wv = (const float*)d_in[5];
  const float* bv = (const float*)d_in[6];
  const float* wo = (const float*)d_in[7];
  const float* bo = (const float*)d_in[8];

  char* ws = (char*)d_ws;
  __hip_bfloat16* xb   = (__hip_bfloat16*)(ws);              // 16 MB  x  bf16
  __hip_bfloat16* qbf  = (__hip_bfloat16*)(ws + 16777216);   // 16 MB  q  (scaled)
  __hip_bfloat16* kbf  = (__hip_bfloat16*)(ws + 33554432);   // 16 MB  k
  __hip_bfloat16* vtb  = (__hip_bfloat16*)(ws + 50331648);   // 16 MB  v transposed
  __hip_bfloat16* ctxb = (__hip_bfloat16*)(ws + 67108864);   // 16 MB  ctx
  __hip_bfloat16* wqb  = (__hip_bfloat16*)(ws + 83886080);   //  2 MB
  __hip_bfloat16* wkb  = (__hip_bfloat16*)(ws + 85983232);
  __hip_bfloat16* wvb  = (__hip_bfloat16*)(ws + 88080384);
  __hip_bfloat16* wob  = (__hip_bfloat16*)(ws + 90177536);   // total 88 MB

  float* out = (float*)d_out;                        // [T,B,E] fp32
  float* avg = out + (size_t)TGT * BSZ * EMB;        // [B,T,T] fp32

  const int M = TGT * BSZ;                           // 8192

  cvt_f32_bf16<<<2048, 256, 0, stream>>>(x,  xb,  M * EMB);
  cvt_f32_bf16<<<512,  256, 0, stream>>>(wq, wqb, EMB * EMB);
  cvt_f32_bf16<<<512,  256, 0, stream>>>(wk, wkb, EMB * EMB);
  cvt_f32_bf16<<<512,  256, 0, stream>>>(wv, wvb, EMB * EMB);
  cvt_f32_bf16<<<512,  256, 0, stream>>>(wo, wob, EMB * EMB);

  dim3 gg(EMB / 128, M / 64);                        // (8,128) blocks, 256 thr
  const float scaling = 0.125f;                      // HD^-0.5
  gemm_wmma<0><<<gg, 256, 0, stream>>>(xb, wqb, bq, qbf, M, EMB, EMB, scaling);
  gemm_wmma<0><<<gg, 256, 0, stream>>>(xb, wkb, bk, kbf, M, EMB, EMB, 1.0f);
  gemm_wmma<1><<<gg, 256, 0, stream>>>(xb, wvb, bv, vtb, M, EMB, EMB, 1.0f);

  const size_t attn_lds = 131072 + (size_t)16 * PLD * 2 + 8192;  // 205312 B
  attn_wmma<<<dim3(BSZ * (TGT / 16)), 256, attn_lds, stream>>>(qbf, kbf, vtb, ctxb, avg);

  gemm_wmma<2><<<gg, 256, 0, stream>>>(ctxb, wob, bo, (void*)out, M, EMB, EMB, 1.0f);
}
